// TransducerModel_18915035972189
// MI455X (gfx1250) — matrix operand order repaired and checked by reading, not compiled, not run on hardware
//
#include <hip/hip_runtime.h>
#include <hip/hip_bf16.h>

typedef _Float16 v16h __attribute__((ext_vector_type(16)));
typedef _Float16 v8h  __attribute__((ext_vector_type(8)));
typedef float    v8f  __attribute__((ext_vector_type(8)));

#define STEPS 257
#define BATCH 32
#define HD    1024
#define H3    3072
#define VOCAB 10001
#define VPAD  10048
#define MROWS (STEPS * BATCH) /* 8224 */

// ---------------- workspace layout (bytes, all 256-aligned) ----------------
constexpr size_t OFF_WIH16  = 0;                          // 2*3072*1024 f16
constexpr size_t OFF_WHH16  = OFF_WIH16  + 12582912ull;   // 2*3072*1024 f16
constexpr size_t OFF_WOUT16 = OFF_WHH16  + 12582912ull;   // 10048*1024 f16
constexpr size_t OFF_X16    = OFF_WOUT16 + 20578304ull;   // 8224*1024 f16
constexpr size_t OFF_GI1    = OFF_X16    + 16842752ull;   // 8224*3072 f16
constexpr size_t OFF_H2ALL  = OFF_GI1    + 50528256ull;   // 8224*1024 f16
constexpr size_t OFF_GH1    = OFF_H2ALL  + 16842752ull;   // 32*3072 f32
constexpr size_t OFF_GH2    = OFF_GH1    + 393216ull;     // 32*3072 f32
constexpr size_t OFF_GI2    = OFF_GH2    + 393216ull;     // 32*3072 f32
constexpr size_t OFF_H1F    = OFF_GI2    + 393216ull;     // 32*1024 f32
constexpr size_t OFF_H2F    = OFF_H1F    + 131072ull;     // 32*1024 f32
constexpr size_t OFF_H1H    = OFF_H2F    + 131072ull;     // 32*1024 f16
constexpr size_t OFF_H2H    = OFF_H1H    + 65536ull;      // 32*1024 f16
constexpr size_t OFF_BAR    = OFF_H2H    + 65536ull;      // 4096 int

// ---------------- WMMA fragment helpers (ISA 7.12.2 layouts) ----------------
// A: 16x32 f16 (MxK), row-major, lda halves. lane<16: row=lane, K 0..7 & 16..23;
// lane>=16: row=lane-16, K 8..15 & 24..31.
__device__ inline v16h load_a_frag(const _Float16* A, int lda) {
  int lane = threadIdx.x & 31;
  const _Float16* p = A + (lane & 15) * lda + ((lane >> 4) << 3);
  union { v16h v; v8h h[2]; } u;
  u.h[0] = *(const v8h*)(p);
  u.h[1] = *(const v8h*)(p + 16);
  return u.v;
}
// B: 32x16 f16 (KxN) taken from weight W stored (N,K) row-major (i.e. B = W^T).
// lane<16: col n=lane, K 0..7 & 16..23; lane>=16: same col, K 8..15 & 24..31.
__device__ inline v16h load_b_frag(const _Float16* W, int ldb) {
  int lane = threadIdx.x & 31;
  const _Float16* p = W + (lane & 15) * ldb + ((lane >> 4) << 3);
  union { v16h v; v8h h[2]; } u;
  u.h[0] = *(const v8h*)(p);
  u.h[1] = *(const v8h*)(p + 16);
  return u.v;
}

// one 16x64 output tile, K = HD, A row-major lda=HD, W rows = 64 output cols
__device__ inline void mma_16x64(const _Float16* A, const _Float16* W, v8f acc[4]) {
  for (int k = 0; k < HD; k += 32) {
    v16h a = load_a_frag(A + k, HD);
    __builtin_prefetch(A + k + 128, 0, 1);
#pragma unroll
    for (int t = 0; t < 4; ++t) {
      v16h b = load_b_frag(W + (size_t)t * 16 * HD + k, HD);
      acc[t] = __builtin_amdgcn_wmma_f32_16x16x32_f16(false, a, false, b,
                                                      (short)0, acc[t], false, false);
    }
  }
}

__device__ inline void store_tile_f32(float* C, int ldc, const float* bias64,
                                      const v8f acc[4]) {
  int lane = threadIdx.x & 31;
  int n = lane & 15, mb = (lane >> 4) << 3;
#pragma unroll
  for (int t = 0; t < 4; ++t) {
    float bv = bias64[t * 16 + n];
#pragma unroll
    for (int r = 0; r < 8; ++r)
      C[(size_t)(mb + r) * ldc + t * 16 + n] = acc[t][r] + bv;
  }
}

__device__ inline float sigmoidf_(float x) { return 1.f / (1.f + __expf(-x)); }

// ---------------- grid barrier (monotonic slot IDs, zeroed each launch) -----
__device__ inline void grid_barrier(int* bar, int id, int nwg) {
  __syncthreads();
  if (threadIdx.x == 0) {
    __hip_atomic_fetch_add(&bar[id], 1, __ATOMIC_ACQ_REL, __HIP_MEMORY_SCOPE_AGENT);
    while (__hip_atomic_load(&bar[id], __ATOMIC_ACQUIRE, __HIP_MEMORY_SCOPE_AGENT) < nwg)
      __builtin_amdgcn_s_sleep(1);
  }
  __syncthreads();
}

// ---------------- kernels ----------------
__global__ void init_kernel(const float* init_state, float* h1f, float* h2f,
                            _Float16* h1h, _Float16* h2h, int* bar) {
  int i = blockIdx.x * blockDim.x + threadIdx.x;
  int stride = gridDim.x * blockDim.x;
  for (int idx = i; idx < 4096; idx += stride) bar[idx] = 0;
  for (int idx = i; idx < BATCH * HD; idx += stride) {
    int j = idx & (HD - 1);
    float a = init_state[j], b = init_state[HD + j];
    h1f[idx] = a; h2f[idx] = b;
    h1h[idx] = (_Float16)a; h2h[idx] = (_Float16)b;
  }
}

__global__ void convert_weights_kernel(const float* wih, const float* whh,
                                       const float* wout, _Float16* wih16,
                                       _Float16* whh16, _Float16* wout16) {
  size_t i = (size_t)blockIdx.x * blockDim.x + threadIdx.x;
  size_t stride = (size_t)gridDim.x * blockDim.x;
  const size_t NW = (size_t)2 * H3 * HD;
  for (size_t idx = i; idx < NW; idx += stride) {
    wih16[idx] = (_Float16)wih[idx];
    whh16[idx] = (_Float16)whh[idx];
  }
  const size_t NO = (size_t)VPAD * HD;
  for (size_t idx = i; idx < NO; idx += stride) {
    size_t row = idx >> 10, col = idx & (HD - 1);
    wout16[idx] = (row < VOCAB) ? (_Float16)wout[row * HD + col] : (_Float16)0.f;
  }
}

__global__ void gather_kernel(const int* y, const float* embed, _Float16* X) {
  size_t i = (size_t)blockIdx.x * blockDim.x + threadIdx.x;
  size_t stride = (size_t)gridDim.x * blockDim.x;
  const size_t N = (size_t)MROWS * HD;
  for (size_t idx = i; idx < N; idx += stride) {
    int row = (int)(idx >> 10), col = (int)(idx & (HD - 1));
    int u = row >> 5, b = row & 31;
    int tok = (u == 0) ? (VOCAB - 1) : y[b * 256 + (u - 1)];
    X[idx] = (_Float16)embed[(size_t)tok * HD + col];
  }
}

// GI1 = X @ Wih0^T + bih0   (8224 x 3072, K=1024), stored f16
__global__ __launch_bounds__(256) void gemm_gi1_kernel(const _Float16* X,
    const _Float16* W, const float* bias, _Float16* GI1) {
  int wave = (blockIdx.x * blockDim.x + threadIdx.x) >> 5;
  int nwaves = (gridDim.x * blockDim.x) >> 5;
  const int NJOBS = (MROWS / 16) * (H3 / 64);
  for (int job = wave; job < NJOBS; job += nwaves) {
    int mt = job / (H3 / 64), n4 = job % (H3 / 64);
    v8f acc[4] = {};
    mma_16x64(X + (size_t)mt * 16 * HD, W + (size_t)n4 * 64 * HD, acc);
    int lane = threadIdx.x & 31;
    int n = lane & 15, mb = (lane >> 4) << 3;
#pragma unroll
    for (int t = 0; t < 4; ++t) {
      float bv = bias[n4 * 64 + t * 16 + n];
#pragma unroll
      for (int r = 0; r < 8; ++r)
        GI1[(size_t)(mt * 16 + mb + r) * H3 + n4 * 64 + t * 16 + n] =
            (_Float16)(acc[t][r] + bv);
    }
  }
}

#define RWG 8
#define RTHREADS 512
// persistent sequential GRU recurrence, weights L2-resident
__global__ __launch_bounds__(RTHREADS) void recurrence_kernel(
    const _Float16* whh16, const float* bhh, const _Float16* wih2,
    const float* bih2, const _Float16* gi1, float* gh1, float* gh2, float* gi2,
    float* h1f, float* h2f, _Float16* h1h, _Float16* h2h, _Float16* h2all,
    int* bar) {
  const int tid = threadIdx.x;
  const int gthread = blockIdx.x * RTHREADS + tid;
  const int wave = gthread >> 5;
  const int nwaves = (RWG * RTHREADS) >> 5;       // 128
  const int nthreads = RWG * RTHREADS;            // 4096

  for (int u = 0; u < STEPS; ++u) {
    // Phase A: gh1 = h1@Whh0^T + bhh0 ; gh2 = h2@Whh1^T + bhh1 (parallel)
    for (int job = wave; job < 192; job += nwaves) {
      int mat = job / 96, rem = job % 96;
      int mt = rem / 48, n4 = rem % 48;
      const _Float16* A = (mat ? h2h : h1h) + (size_t)mt * 16 * HD;
      const _Float16* W = whh16 + (size_t)mat * H3 * HD + (size_t)n4 * 64 * HD;
      v8f acc[4] = {};
      mma_16x64(A, W, acc);
      store_tile_f32((mat ? gh2 : gh1) + (size_t)mt * 16 * H3 + n4 * 64, H3,
                     bhh + mat * H3 + n4 * 64, acc);
    }
    grid_barrier(bar, u * 4 + 0, RWG);
    // Phase B: layer-1 gates
    for (int i = gthread; i < BATCH * HD; i += nthreads) {
      int b = i >> 10, j = i & (HD - 1);
      const _Float16* g = gi1 + (size_t)(u * BATCH + b) * H3;
      float ir = (float)g[j], iz = (float)g[j + HD], in_ = (float)g[j + 2 * HD];
      float hr = gh1[b * H3 + j], hz = gh1[b * H3 + j + HD],
            hn = gh1[b * H3 + j + 2 * HD];
      float h = h1f[i];
      float r = sigmoidf_(ir + hr), z = sigmoidf_(iz + hz);
      float n = tanhf(in_ + r * hn);
      float hnew = (1.f - z) * n + z * h;
      h1f[i] = hnew; h1h[i] = (_Float16)hnew;
    }
    grid_barrier(bar, u * 4 + 1, RWG);
    // Phase C: gi2 = h1_new @ Wih1^T + bih1
    for (int job = wave; job < 96; job += nwaves) {
      int mt = job / 48, n4 = job % 48;
      v8f acc[4] = {};
      mma_16x64(h1h + (size_t)mt * 16 * HD, wih2 + (size_t)n4 * 64 * HD, acc);
      store_tile_f32(gi2 + (size_t)mt * 16 * H3 + n4 * 64, H3, bih2 + n4 * 64, acc);
    }
    grid_barrier(bar, u * 4 + 2, RWG);
    // Phase D: layer-2 gates + record h2
    for (int i = gthread; i < BATCH * HD; i += nthreads) {
      int b = i >> 10, j = i & (HD - 1);
      float ir = gi2[b * H3 + j], iz = gi2[b * H3 + j + HD],
            in_ = gi2[b * H3 + j + 2 * HD];
      float hr = gh2[b * H3 + j], hz = gh2[b * H3 + j + HD],
            hn = gh2[b * H3 + j + 2 * HD];
      float h = h2f[i];
      float r = sigmoidf_(ir + hr), z = sigmoidf_(iz + hz);
      float n = tanhf(in_ + r * hn);
      float hnew = (1.f - z) * n + z * h;
      h2f[i] = hnew; h2h[i] = (_Float16)hnew;
      h2all[(size_t)u * BATCH * HD + i] = (_Float16)hnew;
    }
    grid_barrier(bar, u * 4 + 3, RWG);
  }
}

// OUT[b,u,v] = H2ALL[u*32+b,:] @ Wout^T + bout  (8224 x 10048, K=1024)
__global__ __launch_bounds__(256) void gemm_out_kernel(const _Float16* H2,
    const _Float16* Wo, const float* bout, float* out) {
  int wave = (blockIdx.x * blockDim.x + threadIdx.x) >> 5;
  int nwaves = (gridDim.x * blockDim.x) >> 5;
  const int N4 = VPAD / 64;                       // 157
  const int NJOBS = (MROWS / 16) * N4;            // 80698
  for (int job = wave; job < NJOBS; job += nwaves) {
    int mt = job / N4, n4 = job % N4;
    v8f acc[4] = {};
    mma_16x64(H2 + (size_t)mt * 16 * HD, Wo + (size_t)n4 * 64 * HD, acc);
    int lane = threadIdx.x & 31;
    int ni = lane & 15, mb = (lane >> 4) << 3;
#pragma unroll
    for (int t = 0; t < 4; ++t) {
      int n = n4 * 64 + t * 16 + ni;
      if (n < VOCAB) {
        float bv = bout[n];
#pragma unroll
        for (int r = 0; r < 8; ++r) {
          int row = mt * 16 + mb + r;
          int uu = row >> 5, bb = row & 31;
          out[((size_t)bb * STEPS + uu) * VOCAB + n] = acc[t][r] + bv;
        }
      }
    }
  }
}

// ---------------- host launcher ----------------
extern "C" void kernel_launch(void* const* d_in, const int* in_sizes, int n_in,
                              void* d_out, int out_size, void* d_ws, size_t ws_size,
                              hipStream_t stream) {
  (void)in_sizes; (void)n_in; (void)out_size; (void)ws_size;
  const int*   y          = (const int*)d_in[0];
  const float* embed      = (const float*)d_in[2];
  const float* init_state = (const float*)d_in[3];
  const float* Wih        = (const float*)d_in[4];
  const float* Whh        = (const float*)d_in[5];
  const float* bih        = (const float*)d_in[6];
  const float* bhh        = (const float*)d_in[7];
  const float* Wout       = (const float*)d_in[8];
  const float* bout       = (const float*)d_in[9];
  float* out = (float*)d_out;
  char*  ws  = (char*)d_ws;

  _Float16* wih16  = (_Float16*)(ws + OFF_WIH16);
  _Float16* whh16  = (_Float16*)(ws + OFF_WHH16);
  _Float16* wout16 = (_Float16*)(ws + OFF_WOUT16);
  _Float16* x16    = (_Float16*)(ws + OFF_X16);
  _Float16* gi1    = (_Float16*)(ws + OFF_GI1);
  _Float16* h2all  = (_Float16*)(ws + OFF_H2ALL);
  float*    gh1    = (float*)(ws + OFF_GH1);
  float*    gh2    = (float*)(ws + OFF_GH2);
  float*    gi2    = (float*)(ws + OFF_GI2);
  float*    h1f    = (float*)(ws + OFF_H1F);
  float*    h2f    = (float*)(ws + OFF_H2F);
  _Float16* h1h    = (_Float16*)(ws + OFF_H1H);
  _Float16* h2h    = (_Float16*)(ws + OFF_H2H);
  int*      bar    = (int*)(ws + OFF_BAR);

  init_kernel<<<128, 256, 0, stream>>>(init_state, h1f, h2f, h1h, h2h, bar);
  convert_weights_kernel<<<4096, 256, 0, stream>>>(Wih, Whh, Wout, wih16, whh16, wout16);
  gather_kernel<<<2048, 256, 0, stream>>>(y, embed, x16);
  gemm_gi1_kernel<<<3084, 256, 0, stream>>>(x16, wih16, bih, gi1);
  recurrence_kernel<<<RWG, RTHREADS, 0, stream>>>(
      whh16, bhh, wih16 + (size_t)H3 * HD, bih + H3, gi1, gh1, gh2, gi2,
      h1f, h2f, h1h, h2h, h2all, bar);
  gemm_out_kernel<<<4096, 256, 0, stream>>>(h2all, wout16, bout, out);
}
